// DecouplingSemantic_47081431499171
// MI455X (gfx1250) — matrix operand order repaired and checked by reading, not compiled, not run on hardware
//
#include <hip/hip_runtime.h>
#include <hip/hip_bf16.h>
#include <hip/hip_fp16.h>
#include <stdint.h>

typedef __attribute__((ext_vector_type(16))) _Float16 v16h;
typedef __attribute__((ext_vector_type(8)))  float    v8f;

#define BS   16
#define IMG  196
#define CLS  80
#define DIM  512
#define WDIM 300
#define MT   13   // M tiles of 16 (208 rows, 196 valid + padding)
#define KT   16   // K slices of 32 (512 total)
#define NT   32   // N tiles of 16 (512 total)
#define NWAVE 8   // waves per WG; each wave covers NT/NWAVE = 4 N-tiles

// ---------------------------------------------------------------- utilities
__device__ inline float fast_tanh(float x) {
  // tanh(x) = sign(x) * (1 - e^{-2|x|}) / (1 + e^{-2|x|}); v_exp_f32 trans op
  float t = __expf(-2.0f * fabsf(x));
  float r = (1.0f - t) / (1.0f + t);
  return copysignf(r, x);
}

// ---------------------------------------------------------------- kernels
__global__ void zero_out_kernel(float* __restrict__ out, int n) {
  int i = blockIdx.x * blockDim.x + threadIdx.x;
  if (i < n) out[i] = 0.0f;
}

// Pre-swizzle W3^T into f16 WMMA B-fragment order.
// B (32x16 f16) per v_wmma_f32_16x16x32_f16: v16h element i, lanes 0-15:
//   K = kk*32 + (i<8 ? i : i+8), N = lane; lanes 16-31: +8, N = lane-16.
// Fragment dword index: ((n*KT + kk)*32 + lane)*8 + j, halves (2j, 2j+1).
__global__ void prep_b_kernel(const float* __restrict__ W3, uint32_t* __restrict__ Bh) {
  int dw = blockIdx.x * blockDim.x + threadIdx.x;   // 0 .. 131071
  int j      = dw & 7;
  int lane   = (dw >> 3) & 31;
  int kk     = (dw >> 8) & 15;
  int n      = dw >> 12;
  int laneHi = lane >> 4;
  int e = n * 16 + (lane & 15);                     // output column (W3 row)
  int K = kk * 32 + 8 * laneHi + 2 * j + ((j >= 4) ? 8 : 0);  // contraction index d
  union { _Float16 h[2]; uint32_t u; } cv;
  cv.h[0] = (_Float16)W3[e * DIM + K];
  cv.h[1] = (_Float16)W3[e * DIM + K + 1];
  Bh[dw] = cv.u;
}

// One workgroup per (b, c): fused lab -> tanh-product (full A staged once in
// LDS) -> GEMM(W3) with 4 sequential N-tile passes per wave -> softmax(s)
// -> atomic accumulation over c into out.
__global__ __launch_bounds__(256, 1)
void decoupling_main(const float* __restrict__ token,
                     const float* __restrict__ label,
                     const float* __restrict__ W1,
                     const float* __restrict__ b3,
                     const v16h* __restrict__ Bfrag,
                     float* __restrict__ out) {
  __shared__ float sL[WDIM];            // label row (b,c)            1.2 KB
  __shared__ float sLab[DIM];           // lab[b,c,:]                 2   KB
  __shared__ v16h  sA[KT * MT * 32];    // full A, fragment layout  208   KB

  const int b = blockIdx.x / CLS;
  const int c = blockIdx.x % CLS;
  const int t = threadIdx.x;            // 0..255
  const int lane   = t & 31;
  const int wave   = t >> 5;            // 0..7
  const int laneHi = lane >> 4;

  // ---- stage label row, then lab[d] = sum_w L[w] * W1[d,w] ----
  for (int w = t; w < WDIM; w += 256) sL[w] = label[(b * CLS + c) * WDIM + w];
  __syncthreads();
  for (int d = t; d < DIM; d += 256) {
    const float* wrow = W1 + d * WDIM;
    float a = 0.0f;
    for (int w = 0; w < WDIM; ++w) a += sL[w] * wrow[w];
    sLab[d] = a;
  }
  __syncthreads();

  uint32_t* sAdw = (uint32_t*)sA;
  const float* tokB = token + (size_t)b * IMG * DIM;

  // ---- produce the entire A matrix (tanh(token*lab) as f16 fragments) ----
  // 16-bit A 16x32 layout (ISA 05_wmma): lane<16: elems 0-7 -> K 0-7,
  // elems 8-15 -> K 16-23; lane>=16: +8.  Dword j holds elems (2j, 2j+1).
  for (int kk = 0; kk < KT; ++kk) {
    const int kbase = kk * 32;
    for (int idx = t; idx < MT * 256; idx += 256) {
      int mt    = idx >> 8;
      int rem   = idx & 255;
      int alane = rem >> 3;
      int j     = rem & 7;
      int srow  = mt * 16 + (alane & 15);
      int aHi   = alane >> 4;
      int i0    = 2 * j;
      int k     = kbase + i0 + 8 * aHi + ((i0 & 8) ? 8 : 0);
      uint32_t packed = 0u;   // zero-pad rows s >= 196
      if (srow < IMG) {
        const float* tr = tokB + srow * DIM + k;
        union { _Float16 h[2]; uint32_t u; } cv;
        cv.h[0] = (_Float16)fast_tanh(tr[0] * sLab[k]);
        cv.h[1] = (_Float16)fast_tanh(tr[1] * sLab[k + 1]);
        packed = cv.u;
      }
      sAdw[kk * (MT * 256) + idx] = packed;
    }
  }
  __syncthreads();   // A complete; no more barriers needed below

  // ---- 4 N-tile passes per wave ----
  for (int pass = 0; pass < NT / NWAVE; ++pass) {
    const int n = wave * (NT / NWAVE) + pass;

    v8f acc[MT] = {};
    for (int kk = 0; kk < KT; ++kk) {
      v16h bf = Bfrag[(n * KT + kk) * 32 + lane];
#pragma unroll
      for (int m = 0; m < MT; ++m) {
        v16h af = sA[(kk * MT + m) * 32 + lane];
#ifdef __HIP_DEVICE_COMPILE__
        acc[m] = __builtin_amdgcn_wmma_f32_16x16x32_f16(
            /*neg_a=*/false, af, /*neg_b=*/false, bf,
            /*c_mod=*/(short)0, acc[m], /*reuse_a=*/false, /*reuse_b=*/false);
#endif
      }
    }

    // ---- epilogue: + b3, mask padding, softmax over s, weighted output ----
    // C layout: VGPR j, lanes 0-15: M = j, N = lane; lanes 16-31: M = j+8.
    const int   e    = n * 16 + (lane & 15);
    const float bias = b3[e];

    float mx = -INFINITY;
#pragma unroll
    for (int m = 0; m < MT; ++m) {
#pragma unroll
      for (int j = 0; j < 8; ++j) {
        int srow = m * 16 + j + 8 * laneHi;
        float v = (srow < IMG) ? (acc[m][j] + bias) : -INFINITY;
        acc[m][j] = v;
        mx = fmaxf(mx, v);
      }
    }
    mx = fmaxf(mx, __shfl_xor(mx, 16, 32));  // lane pair (l, l^16) shares column e

    float sum = 0.0f;
#pragma unroll
    for (int m = 0; m < MT; ++m) {
#pragma unroll
      for (int j = 0; j < 8; ++j) {
        float p = __expf(acc[m][j] - mx);    // exp(-inf) == 0 for padded rows
        acc[m][j] = p;
        sum += p;
      }
    }
    sum += __shfl_xor(sum, 16, 32);
    const float inv = 1.0f / sum;

#pragma unroll
    for (int m = 0; m < MT; ++m) {
#pragma unroll
      for (int j = 0; j < 8; ++j) {
        int srow = m * 16 + j + 8 * laneHi;
        if (srow < IMG) {
          size_t oidx = ((size_t)b * IMG + srow) * DIM + e;
          float tok = tokB[srow * DIM + e];
          unsafeAtomicAdd(&out[oidx], tok * acc[m][j] * inv);  // global_atomic_add_f32
        }
      }
    }
  }
}

// ---------------------------------------------------------------- launcher
extern "C" void kernel_launch(void* const* d_in, const int* in_sizes, int n_in,
                              void* d_out, int out_size, void* d_ws, size_t ws_size,
                              hipStream_t stream) {
  const float* token = (const float*)d_in[0];  // (16,196,512)
  const float* label = (const float*)d_in[1];  // (16,80,300)
  const float* W1    = (const float*)d_in[2];  // (512,300)
  const float* W3    = (const float*)d_in[3];  // (512,512)
  const float* b3    = (const float*)d_in[4];  // (512,)
  float* out = (float*)d_out;                  // (16,196,512)

  uint32_t* Bh = (uint32_t*)d_ws;              // 512 KB f16 B-fragment table

  zero_out_kernel<<<(out_size + 1023) / 1024, 1024, 0, stream>>>(out, out_size);
  prep_b_kernel<<<(NT * KT * 32 * 8) / 256, 256, 0, stream>>>(W3, Bh);
  decoupling_main<<<BS * CLS, 256, 0, stream>>>(token, label, W1, b3,
                                                (const v16h*)d_ws, out);
}
